// SelfAttention2d_2035814498328
// MI455X (gfx1250) — compile-verified, matrix operand order checked by reading, not run on hardware
//
#include <hip/hip_runtime.h>
#include <hip/hip_bf16.h>
#include <stdint.h>

// ---------------------------------------------------------------------------
// Fused self-attention for [B=8, C=64, H=64, W=64], N = 4096, C' = 8.
//   Pass 0: q/k/v 1x1-conv projections (VALU, weights staged in LDS)
//   Pass A: r[n] = sum_j exp(k[:,n]·q[:,j])     via V_WMMA_F32_16X16X4_F32
//   Pass B: out  = (V * diag(1/r)) @ exp(K^T Q) via V_WMMA_F32_16X16X32_BF16
// Attention matrix is never materialized (saves ~1 GB of HBM traffic).
// ---------------------------------------------------------------------------

typedef __attribute__((ext_vector_type(2)))  float  v2f;
typedef __attribute__((ext_vector_type(8)))  float  v8f;
typedef __attribute__((ext_vector_type(16))) __bf16 v16bf;

#define NPOS 4096
#define CCH  64
#define CQK  8
#define NBAT 8

union BF16x16 { v16bf h; uint32_t u[8]; };

__device__ __forceinline__ v8f wmma_f32_4(v2f a, v2f b, v8f c) {
  // D = A(16x4 f32) * B(4x16 f32) + C
  return __builtin_amdgcn_wmma_f32_16x16x4_f32(false, a, false, b,
                                               (short)0, c, false, false);
}

// ---------------- Pass 0: QKV projection -----------------------------------
__global__ __launch_bounds__(256) void qkv_proj(
    const float* __restrict__ x,
    const float* __restrict__ wq, const float* __restrict__ bq,
    const float* __restrict__ wk, const float* __restrict__ bk,
    const float* __restrict__ wv, const float* __restrict__ bv,
    float* __restrict__ qbuf, float* __restrict__ kbuf,
    float* __restrict__ vbuf) {
  __shared__ float sWq[CQK * CCH], sWk[CQK * CCH], sWv[CCH * CCH];
  __shared__ float sBq[CQK], sBk[CQK], sBv[CCH];
  const int tid = threadIdx.x;
  for (int i = tid; i < CQK * CCH; i += 256) { sWq[i] = wq[i]; sWk[i] = wk[i]; }
  for (int i = tid; i < CCH * CCH; i += 256) sWv[i] = wv[i];
  if (tid < CQK) { sBq[tid] = bq[tid]; sBk[tid] = bk[tid]; }
  if (tid < CCH) sBv[tid] = bv[tid];
  __syncthreads();

  const int idx = blockIdx.x * 256 + tid;     // one (b, n) per thread
  const int b = idx >> 12;
  const int n = idx & (NPOS - 1);
  const float* xb = x + ((size_t)b * CCH) * NPOS + n;

  float xr[CCH];
  #pragma unroll
  for (int c = 0; c < CCH; ++c) xr[c] = xb[(size_t)c * NPOS];   // coalesced in n

  float qa[CQK], ka[CQK];
  #pragma unroll
  for (int o = 0; o < CQK; ++o) { qa[o] = sBq[o]; ka[o] = sBk[o]; }
  #pragma unroll
  for (int c = 0; c < CCH; ++c) {
    const float xv = xr[c];
    #pragma unroll
    for (int o = 0; o < CQK; ++o) {
      qa[o] = fmaf(sWq[o * CCH + c], xv, qa[o]);   // LDS broadcast reads
      ka[o] = fmaf(sWk[o * CCH + c], xv, ka[o]);
    }
  }
  #pragma unroll
  for (int o = 0; o < CQK; ++o) {
    qbuf[((size_t)b * CQK + o) * NPOS + n] = qa[o];
    kbuf[((size_t)b * CQK + o) * NPOS + n] = ka[o];
  }
  for (int ch = 0; ch < 4; ++ch) {             // v in 4 chunks of 16 channels
    float va[16];
    #pragma unroll
    for (int o = 0; o < 16; ++o) va[o] = sBv[ch * 16 + o];
    #pragma unroll
    for (int c = 0; c < CCH; ++c) {
      const float xv = xr[c];
      #pragma unroll
      for (int o = 0; o < 16; ++o)
        va[o] = fmaf(sWv[(ch * 16 + o) * CCH + c], xv, va[o]);
    }
    #pragma unroll
    for (int o = 0; o < 16; ++o)
      vbuf[((size_t)b * CCH + ch * 16 + o) * NPOS + n] = va[o];
  }
}

// ---------------- Pass A: softmax row sums r[n] ----------------------------
// One wave per 16-row n-block; A = K^T tile (fixed), sweep B = Q over all j.
__global__ __launch_bounds__(256) void row_sums(
    const float* __restrict__ qbuf, const float* __restrict__ kbuf,
    float* __restrict__ rinv) {
  const int w    = blockIdx.x * 8 + (threadIdx.x >> 5);
  const int lane = threadIdx.x & 31;
  const int b  = w >> 8;
  const int nb = w & 255;
  const int m    = lane & 15;
  const int ch0  = (lane < 16) ? 0 : 2;       // f32 A/B fragment: K split by half-wave
  const float* kb = kbuf + (size_t)b * CQK * NPOS;
  const float* qb = qbuf + (size_t)b * CQK * NPOS;

  const int n = nb * 16 + m;
  const v2f a_lo = { kb[(ch0 + 0) * NPOS + n], kb[(ch0 + 1) * NPOS + n] };
  const v2f a_hi = { kb[(ch0 + 4) * NPOS + n], kb[(ch0 + 5) * NPOS + n] };

  float rs[8];
  #pragma unroll
  for (int i = 0; i < 8; ++i) rs[i] = 0.0f;

  for (int jt = 0; jt < 256; ++jt) {
    const int j = jt * 16 + m;
    const v2f b_lo = { qb[(ch0 + 0) * NPOS + j], qb[(ch0 + 1) * NPOS + j] };
    const v2f b_hi = { qb[(ch0 + 4) * NPOS + j], qb[(ch0 + 5) * NPOS + j] };
    v8f c = {};
    c = wmma_f32_4(a_lo, b_lo, c);            // channels 0..3
    c = wmma_f32_4(a_hi, b_hi, c);            // channels 4..7
    #pragma unroll
    for (int i = 0; i < 8; ++i) rs[i] += __expf(c[i]);
  }
  // Reduce over the 16 lanes of each half (columns j); rows m live in VGPR idx.
  #pragma unroll
  for (int i = 0; i < 8; ++i) {
    float s = rs[i];
    s += __shfl_xor(s, 8, 16);
    s += __shfl_xor(s, 4, 16);
    s += __shfl_xor(s, 2, 16);
    s += __shfl_xor(s, 1, 16);
    rs[i] = s;
  }
  if (lane == 0 || lane == 16) {
    const int rowbase = nb * 16 + ((lane == 16) ? 8 : 0);
    #pragma unroll
    for (int i = 0; i < 8; ++i)
      rinv[(size_t)b * NPOS + rowbase + i] = 1.0f / rs[i];
  }
}

// ---------------- Pass A.5: V' = V * diag(1/r) in bf16 ---------------------
__global__ __launch_bounds__(256) void scale_v(
    const float* __restrict__ vbuf, const float* __restrict__ rinv,
    uint16_t* __restrict__ vprime) {
  const size_t idx = (size_t)blockIdx.x * 256 + threadIdx.x;
  const int n = (int)(idx & (NPOS - 1));
  const int b = (int)(idx >> 18);              // idx / (64*4096)
  const float f = vbuf[idx] * rinv[(size_t)b * NPOS + n];
  const uint32_t u = __float_as_uint(f);
  vprime[idx] = (uint16_t)((u + 0x7FFFu + ((u >> 16) & 1u)) >> 16);  // RNE bf16
}

// ---------------- Pass B: out = V' @ exp(K^T Q), + residual ----------------
// One wave per (batch, 16-wide j block); produces all 64 output channels.
__global__ __launch_bounds__(256) void attn_out(
    const float* __restrict__ qbuf, const float* __restrict__ kbuf,
    const uint16_t* __restrict__ vprime, const float* __restrict__ x,
    const float* __restrict__ gamma, float* __restrict__ out) {
  const int w    = blockIdx.x * 8 + (threadIdx.x >> 5);
  const int lane = threadIdx.x & 31;
  const int b  = w >> 8;
  const int jb = w & 255;
  const int m    = lane & 15;
  const int half = lane >> 4;
  const int ch0  = half * 2;

  const float*    qb = qbuf   + (size_t)b * CQK * NPOS;
  const float*    kb = kbuf   + (size_t)b * CQK * NPOS;
  const uint16_t* vp = vprime + (size_t)b * CCH * NPOS;

  const int j = jb * 16 + m;
  const v2f q_lo = { qb[(ch0 + 0) * NPOS + j], qb[(ch0 + 1) * NPOS + j] };
  const v2f q_hi = { qb[(ch0 + 4) * NPOS + j], qb[(ch0 + 5) * NPOS + j] };

  v8f acc[4];
  #pragma unroll
  for (int cb = 0; cb < 4; ++cb) acc[cb] = (v8f){};

  for (int n0 = 0; n0 < NPOS; n0 += 32) {
    if (n0 + 32 < NPOS)                        // pull next V' lines into cache
      __builtin_prefetch(vp + (size_t)m * NPOS + n0 + 32, 0, 1);

    // Recompute two 16x16 exp(S) tiles (rows n0..n0+31) with f32 WMMA.
    v8f e[2];
    #pragma unroll
    for (int t = 0; t < 2; ++t) {
      const int n = n0 + t * 16 + m;
      const v2f a_lo = { kb[(ch0 + 0) * NPOS + n], kb[(ch0 + 1) * NPOS + n] };
      const v2f a_hi = { kb[(ch0 + 4) * NPOS + n], kb[(ch0 + 5) * NPOS + n] };
      v8f c = {};
      c = wmma_f32_4(a_lo, q_lo, c);
      c = wmma_f32_4(a_hi, q_hi, c);
      #pragma unroll
      for (int i = 0; i < 8; ++i) e[t][i] = __expf(c[i]);
    }
    // C-layout (row = vgpr + 8*half, col = lane&15) maps onto the bf16
    // B-operand 32x16 fragment (Kpair = 2*vgpr + 8*half) by per-lane packing:
    BF16x16 bs;
    #pragma unroll
    for (int i = 0; i < 8; ++i) {
      bs.h[i]     = (__bf16)e[0][i];
      bs.h[8 + i] = (__bf16)e[1][i];
    }
    // 64 output channels = 4 bf16 WMMAs sharing the B fragment.
    #pragma unroll
    for (int cb = 0; cb < 4; ++cb) {
      BF16x16 av;
      #pragma unroll
      for (int r = 0; r < 8; ++r) {
        const int kk = ((r < 4) ? 2 * r : 16 + 2 * (r - 4)) + half * 8;
        av.u[r] = *(const uint32_t*)(vp + (size_t)(cb * 16 + m) * NPOS + n0 + kk);
      }
      acc[cb] = __builtin_amdgcn_wmma_f32_16x16x32_bf16(
          false, av.h, false, bs.h, (short)0, acc[cb], false, false);
    }
  }

  const float g = gamma[0];
  const float* xb = x   + (size_t)b * CCH * NPOS;
  float*       ob = out + (size_t)b * CCH * NPOS;
  #pragma unroll
  for (int cb = 0; cb < 4; ++cb) {
    #pragma unroll
    for (int r = 0; r < 8; ++r) {
      const int c = cb * 16 + r + half * 8;    // C/D tile row mapping
      const size_t o = (size_t)c * NPOS + j;
      ob[o] = fmaf(g, acc[cb][r], xb[o]);
    }
  }
}

// ---------------------------------------------------------------------------
extern "C" void kernel_launch(void* const* d_in, const int* in_sizes, int n_in,
                              void* d_out, int out_size, void* d_ws, size_t ws_size,
                              hipStream_t stream) {
  const float* x     = (const float*)d_in[0];
  const float* wq    = (const float*)d_in[1];
  const float* bq    = (const float*)d_in[2];
  const float* wk    = (const float*)d_in[3];
  const float* bk    = (const float*)d_in[4];
  const float* wv    = (const float*)d_in[5];
  const float* bv    = (const float*)d_in[6];
  const float* gamma = (const float*)d_in[7];
  float* out = (float*)d_out;

  // Workspace layout (bytes):
  //   qbuf  f32 [8][8][4096]   @ 0          (1 MiB)
  //   kbuf  f32 [8][8][4096]   @ 1 MiB      (1 MiB)
  //   vbuf  f32 [8][64][4096]  @ 2 MiB      (8 MiB)
  //   rinv  f32 [8][4096]      @ 10 MiB     (128 KiB)
  //   vprim bf16[8][64][4096]  @ 10.125 MiB (4 MiB)   total ~14.2 MiB
  char* ws = (char*)d_ws;
  float*    qbuf   = (float*)(ws);
  float*    kbuf   = (float*)(ws + (1u << 20));
  float*    vbuf   = (float*)(ws + (2u << 20));
  float*    rinv   = (float*)(ws + (10u << 20));
  uint16_t* vprime = (uint16_t*)(ws + (10u << 20) + (1u << 17));

  qkv_proj<<<128, 256, 0, stream>>>(x, wq, bq, wk, bk, wv, bv, qbuf, kbuf, vbuf);
  row_sums<<<256, 256, 0, stream>>>(qbuf, kbuf, rinv);
  scale_v <<<8192, 256, 0, stream>>>(vbuf, rinv, vprime);
  attn_out<<<256, 256, 0, stream>>>(qbuf, kbuf, vprime, x, gamma, out);
}
